// SwinTransformerEncoder_88347477279449
// MI455X (gfx1250) — matrix operand order repaired and checked by reading, not compiled, not run on hardware
//
#include <hip/hip_runtime.h>

// ---------------------------------------------------------------------------
// Transformer encoder layer for MI455X (gfx1250): bf16 WMMA GEMMs, f32 math
// for LN / softmax(query-axis) / GELU epilogues. Weights are repacked
// transposed (bf16 [N][K]) so all weight-GEMM LDS staging is ds_store_b128.
// ---------------------------------------------------------------------------

typedef __attribute__((ext_vector_type(16))) __bf16 v16bf;
typedef __attribute__((ext_vector_type(8)))  float  v8f;

#define TILE_M 128
#define TILE_K 32

__device__ __forceinline__ unsigned short f32_to_bf16(float f) {
    union { float f; unsigned int u; } c; c.f = f;
    unsigned int u = c.u;
    unsigned int r = u + 0x7FFFu + ((u >> 16) & 1u);   // round-to-nearest-even
    return (unsigned short)(r >> 16);
}
__device__ __forceinline__ float bf16_to_f32(unsigned short h) {
    union { unsigned int u; float f; } c; c.u = ((unsigned int)h) << 16;
    return c.f;
}

union FragBf { v16bf v; uint4 q[2]; };

// A fragment (16x32 bf16): lane<16 holds K 0..15; lane>=16
// holds K 16..31 for row M=lane%16.
__device__ __forceinline__ void load_fragA(FragBf& f, const unsigned short* row, int lhalf) {
    const uint4* p = (const uint4*)(row + lhalf * 16);
    f.q[0] = p[0];        // K offsets [k0+16h, k0+16h+8)
    f.q[1] = p[1];        // K offsets [k0+16h+8, k0+16h+16)
}
// B fragment (32x16 bf16): lane<16 -> K 0..15, lane>=16 -> K 16..31, N=lane%16.
__device__ __forceinline__ void load_fragB(FragBf& f, const unsigned short* row, int lhalf) {
    const uint4* p = (const uint4*)(row + lhalf * 16);
    f.q[0] = p[0];
    f.q[1] = p[1];
}

__device__ __forceinline__ float gelu_exact(float x) {
    return 0.5f * x * (1.0f + erff(x * 0.70710678118654752f));
}

// EPI: 0 = bias only, 1 = bias + residual add, 2 = bias + exact GELU
// TRANSB: B stored [N][K] (row stride ldb over K) -> vectorized staging
// TN: workgroup tile N width (64 or 128); M tile fixed at 128, K step 32.
template <int EPI, bool TRANSB, bool OUT_BF16, int TN>
__global__ __launch_bounds__(256) void gemm_wmma_bf16(
    const unsigned short* __restrict__ Abase, int lda, long long sAb, long long sAh,
    const unsigned short* __restrict__ Bbase, int ldb, long long sBb, long long sBh,
    void* __restrict__ Cbase, int ldc, long long sCb, long long sCh,
    const float* __restrict__ bias,
    const float* __restrict__ Rbase, int ldr,
    int K, float alpha, int Hdiv)
{
    constexpr int JF = TN / 32;                 // B fragments per wave (cols/16 per wave half)
    __shared__ alignas(16) unsigned short As[TILE_M][TILE_K]; // 8 KB
    __shared__ alignas(16) unsigned short Bs[TN][TILE_K];     // 4 or 8 KB ([n][k])

    const int z  = blockIdx.z;
    const int zb = z / Hdiv, zh = z - zb * Hdiv;
    const unsigned short* A = Abase + zb * sAb + zh * sAh;
    const unsigned short* B = Bbase + zb * sBb + zh * sBh;
    const long long coff = zb * sCb + zh * sCh;

    const int m0 = blockIdx.x * TILE_M;
    const int n0 = blockIdx.y * TN;
    const int tid = threadIdx.x;
    const int wave = tid >> 5, lane = tid & 31;
    const int wm = wave >> 1, wn = wave & 1;       // 4x2 wave grid
    const int lhalf = lane >> 4, lm = lane & 15;

    v8f acc[2][JF] = {};

    for (int k0 = 0; k0 < K; k0 += TILE_K) {
        // --- stage A: 128x32 bf16 (each thread: two 16B loads) ---
        {
            int r = tid >> 2;                 // 0..63
            int c = (tid & 3) * 8;            // 0,8,16,24
            *(uint4*)&As[r][c]      = *(const uint4*)(A + (long long)(m0 + r) * lda + k0 + c);
            *(uint4*)&As[64 + r][c] = *(const uint4*)(A + (long long)(m0 + 64 + r) * lda + k0 + c);
        }
        // --- stage B into [n][k] layout ---
        if (TRANSB) {
            if (TN == 128) {
                int n = tid >> 1;             // 0..127
                int c = (tid & 1) * 16;       // 0,16
                const unsigned short* src = B + (long long)(n0 + n) * ldb + k0 + c;
                *(uint4*)&Bs[n][c]     = *(const uint4*)(src);
                *(uint4*)&Bs[n][c + 8] = *(const uint4*)(src + 8);
            } else {
                int n = tid >> 2;             // 0..63
                int c = (tid & 3) * 8;        // 0..24
                *(uint4*)&Bs[n][c] = *(const uint4*)(B + (long long)(n0 + n) * ldb + k0 + c);
            }
        } else {
            // row-major B [K][N]; transpose while staging (TN==64 path: attn@V)
            int kk = tid >> 3;                // 0..31
            int n  = (tid & 7) * (TN / 8);    // stride TN/8 cols per thread
            const unsigned short* src = B + (long long)(k0 + kk) * ldb + n0 + n;
            #pragma unroll
            for (int j = 0; j < TN / 8; ++j) Bs[n + j][kk] = src[j];
        }
        __syncthreads();

        FragBf a[2], b[JF];
        #pragma unroll
        for (int i = 0; i < 2; ++i)
            load_fragA(a[i], &As[wm * 32 + i * 16 + lm][0], lhalf);
        #pragma unroll
        for (int j = 0; j < JF; ++j)
            load_fragB(b[j], &Bs[wn * (TN / 2) + j * 16 + lm][0], lhalf);

        #pragma unroll
        for (int i = 0; i < 2; ++i)
            #pragma unroll
            for (int j = 0; j < JF; ++j)
                acc[i][j] = __builtin_amdgcn_wmma_f32_16x16x32_bf16(
                    false, a[i].v, false, b[j].v, (short)0, acc[i][j], false, false);
        __syncthreads();
    }

    // --- epilogue: C/D layout -> VGPR r holds row (r + lhalf*8), col = lane%16
    #pragma unroll
    for (int i = 0; i < 2; ++i) {
        #pragma unroll
        for (int j = 0; j < JF; ++j) {
            const int gmb = m0 + wm * 32 + i * 16 + lhalf * 8;
            const int gn  = n0 + wn * (TN / 2) + j * 16 + lm;
            const float bcol = bias ? bias[gn] : 0.0f;
            #pragma unroll
            for (int r = 0; r < 8; ++r) {
                const int gm = gmb + r;
                float v = acc[i][j][r] * alpha + bcol;
                if (EPI == 1) v += Rbase[(long long)gm * ldr + gn];
                if (EPI == 2) v = gelu_exact(v);
                if (OUT_BF16)
                    ((unsigned short*)Cbase)[coff + (long long)gm * ldc + gn] = f32_to_bf16(v);
                else
                    ((float*)Cbase)[coff + (long long)gm * ldc + gn] = v;
            }
        }
    }
}

// ---------------------------------------------------------------------------
// LayerNorm (one row per block), f32 in -> bf16 out
// ---------------------------------------------------------------------------
__global__ __launch_bounds__(256) void layernorm_bf16(
    const float* __restrict__ x, const float* __restrict__ g, const float* __restrict__ b,
    unsigned short* __restrict__ out, int D)
{
    const long long row = blockIdx.x;
    const float* xr = x + row * D;
    float s = 0.f, s2 = 0.f;
    for (int i = threadIdx.x; i < D; i += 256) { float v = xr[i]; s += v; s2 += v * v; }
    #pragma unroll
    for (int off = 16; off > 0; off >>= 1) {
        s  += __shfl_down(s, off, 32);
        s2 += __shfl_down(s2, off, 32);
    }
    __shared__ float ls[8], ls2[8];
    const int wave = threadIdx.x >> 5, lane = threadIdx.x & 31;
    if (lane == 0) { ls[wave] = s; ls2[wave] = s2; }
    __syncthreads();
    if (threadIdx.x == 0) {
        float a = 0.f, a2 = 0.f;
        for (int i = 0; i < 8; ++i) { a += ls[i]; a2 += ls2[i]; }
        ls[0] = a; ls2[0] = a2;
    }
    __syncthreads();
    const float mean = ls[0] / (float)D;
    const float var  = ls2[0] / (float)D - mean * mean;
    const float inv  = rsqrtf(var + 1e-5f);
    for (int i = threadIdx.x; i < D; i += 256) {
        float v = (xr[i] - mean) * inv * g[i] + b[i];
        out[row * D + i] = f32_to_bf16(v);
    }
}

// ---------------------------------------------------------------------------
// Softmax over the QUERY axis (axis=2 of [BH, Sq, Sk]): per-column max/sum,
// normalize in place (bf16 scores). Coalesced: consecutive threads = cols.
// ---------------------------------------------------------------------------
__global__ __launch_bounds__(256) void softmax_qaxis(unsigned short* __restrict__ sc, int S)
{
    const long long bh = blockIdx.y;
    const int k = blockIdx.x * 256 + threadIdx.x;
    unsigned short* base = sc + bh * (long long)S * S + k;
    float mx = -3.0e38f;
    for (int q = 0; q < S; ++q) mx = fmaxf(mx, bf16_to_f32(base[(long long)q * S]));
    float sum = 0.f;
    for (int q = 0; q < S; ++q) sum += __expf(bf16_to_f32(base[(long long)q * S]) - mx);
    const float inv = 1.0f / sum;
    for (int q = 0; q < S; ++q) {
        float v = __expf(bf16_to_f32(base[(long long)q * S]) - mx) * inv;
        base[(long long)q * S] = f32_to_bf16(v);
    }
}

// ---------------------------------------------------------------------------
// Weight packing / conversion (all weights packed TRANSPOSED: bf16 [N][K])
// ---------------------------------------------------------------------------
// wq/wk/wv [H=16, D=1024, E=64] f32 -> wcatT bf16 [3072][1024] with rows =
// blk*1024 + h*64 + e (blk: 0=q,1=k,2=v). Bias packed in the same col order.
__global__ __launch_bounds__(256) void pack_wqkv_T(
    const float* __restrict__ wq, const float* __restrict__ wk, const float* __restrict__ wv,
    const float* __restrict__ bq, const float* __restrict__ bk, const float* __restrict__ bv,
    unsigned short* __restrict__ wcatT, float* __restrict__ bcat)
{
    const long long idx = (long long)blockIdx.x * 256 + threadIdx.x;
    if (idx >= (long long)3072 * 1024) return;
    const int c = (int)(idx / 1024);   // output row (N index)
    const int d = (int)(idx % 1024);   // output col (K index)
    const int blk = c / 1024 == 0 ? c / 1024 : c / 1024;    // keep simple below
    const int blk3 = c >> 10;                                // 0=q,1=k,2=v
    const int he = c & 1023, h = he >> 6, e = he & 63;
    const float* w = (blk3 == 0) ? wq : (blk3 == 1) ? wk : wv;
    (void)blk;
    wcatT[idx] = f32_to_bf16(w[((long long)h * 1024 + d) * 64 + e]);
    if (d == 0) {
        const float* bb = (blk3 == 0) ? bq : (blk3 == 1) ? bk : bv;
        bcat[c] = bb[he];
    }
}

// out[n*K + k] = bf16(in[k*N + n])  (transpose + convert)
__global__ __launch_bounds__(256) void cvt_bf16_T(
    const float* __restrict__ in, unsigned short* __restrict__ out, int N, int K)
{
    const long long i = (long long)blockIdx.x * 256 + threadIdx.x;
    if (i >= (long long)N * K) return;
    const int n = (int)(i / K), k = (int)(i % K);
    out[i] = f32_to_bf16(in[(long long)k * N + n]);
}

// ---------------------------------------------------------------------------
// Host launcher
// ---------------------------------------------------------------------------
extern "C" void kernel_launch(void* const* d_in, const int* in_sizes, int n_in,
                              void* d_out, int out_size, void* d_ws, size_t ws_size,
                              hipStream_t stream)
{
    (void)in_sizes; (void)n_in; (void)out_size; (void)ws_size;
    constexpr int Bsz = 2, S = 2048, D = 1024, H = 16, FF = 4096, E = 64;
    constexpr int BS = Bsz * S;                 // 4096 rows
    constexpr int NQKV = 3 * H * E;             // 3072

    const float* inputs = (const float*)d_in[0];
    const float* ln1_g  = (const float*)d_in[1];
    const float* ln1_b  = (const float*)d_in[2];
    const float* wq     = (const float*)d_in[3];
    const float* bq     = (const float*)d_in[4];
    const float* wk     = (const float*)d_in[5];
    const float* bk     = (const float*)d_in[6];
    const float* wv     = (const float*)d_in[7];
    const float* bv     = (const float*)d_in[8];
    const float* wo     = (const float*)d_in[9];
    const float* bo     = (const float*)d_in[10];
    const float* ln2_g  = (const float*)d_in[11];
    const float* ln2_b  = (const float*)d_in[12];
    const float* w1     = (const float*)d_in[13];
    const float* b1     = (const float*)d_in[14];
    const float* w2     = (const float*)d_in[15];
    const float* b2     = (const float*)d_in[16];
    float* out = (float*)d_out;

    char* wsp = (char*)d_ws;
    size_t off = 0;
    auto take = [&](size_t bytes) -> char* {
        char* p = wsp + off;
        off += (bytes + 255) & ~(size_t)255;
        return p;
    };
    unsigned short* x1bf   = (unsigned short*)take((size_t)BS * D * 2);
    unsigned short* qkvbf  = (unsigned short*)take((size_t)BS * NQKV * 2);
    unsigned short* scores = (unsigned short*)take((size_t)Bsz * H * S * S * 2);
    unsigned short* abf    = (unsigned short*)take((size_t)BS * D * 2);
    float*          inter  = (float*)take((size_t)BS * D * 4);
    unsigned short* x2bf   = (unsigned short*)take((size_t)BS * D * 2);
    unsigned short* hbf    = (unsigned short*)take((size_t)BS * FF * 2);
    unsigned short* wqkvT  = (unsigned short*)take((size_t)NQKV * D * 2);  // [3072][1024]
    unsigned short* woT    = (unsigned short*)take((size_t)D * D * 2);     // [1024][1024]
    unsigned short* w1T    = (unsigned short*)take((size_t)FF * D * 2);    // [4096][1024]
    unsigned short* w2T    = (unsigned short*)take((size_t)D * FF * 2);    // [1024][4096]
    float*          bqkv   = (float*)take((size_t)NQKV * 4);

    // --- weight conversion (transposed bf16 packing) ---
    pack_wqkv_T<<<(NQKV * D + 255) / 256, 256, 0, stream>>>(wq, wk, wv, bq, bk, bv, wqkvT, bqkv);
    cvt_bf16_T<<<((long long)D * D + 255) / 256, 256, 0, stream>>>(wo, woT, D, D);
    cvt_bf16_T<<<((long long)FF * D + 255) / 256, 256, 0, stream>>>(w1, w1T, FF, D);
    cvt_bf16_T<<<((long long)D * FF + 255) / 256, 256, 0, stream>>>(w2, w2T, D, FF);

    // --- LN1 ---
    layernorm_bf16<<<BS, 256, 0, stream>>>(inputs, ln1_g, ln1_b, x1bf, D);

    // --- fused QKV projection: [4096,1024] x [1024,3072] + bias -> bf16 ---
    gemm_wmma_bf16<0, true, true, 128><<<dim3(BS / TILE_M, NQKV / 128, 1), 256, 0, stream>>>(
        x1bf, D, 0, 0,  wqkvT, D, 0, 0,  qkvbf, NQKV, 0, 0,
        bqkv, nullptr, 0, D, 1.0f, 1);

    // --- scores = Q K^T / sqrt(D): per (b,h): [2048,64] x [64,2048]^T -> bf16 ---
    gemm_wmma_bf16<0, true, true, 128><<<dim3(S / TILE_M, S / 128, Bsz * H), 256, 0, stream>>>(
        qkvbf,          NQKV, (long long)S * NQKV, (long long)E,
        qkvbf + H * E,  NQKV, (long long)S * NQKV, (long long)E,
        scores, S, (long long)H * S * S, (long long)S * S,
        nullptr, nullptr, 0, E, 0.03125f /* 1/sqrt(1024) */, H);

    // --- softmax over query axis (in place, bf16) ---
    softmax_qaxis<<<dim3(S / 256, Bsz * H), 256, 0, stream>>>(scores, S);

    // --- A = attn @ V: per (b,h): [2048,2048] x [2048,64] -> head-concat bf16 ---
    gemm_wmma_bf16<0, false, true, 64><<<dim3(S / TILE_M, 1, Bsz * H), 256, 0, stream>>>(
        scores, S, (long long)H * S * S, (long long)S * S,
        qkvbf + 2 * H * E, NQKV, (long long)S * NQKV, (long long)E,
        abf, D, (long long)S * D, (long long)E,
        nullptr, nullptr, 0, S, 1.0f, H);

    // --- out proj + bias + residual(inputs) -> inter (f32) ---
    gemm_wmma_bf16<1, true, false, 128><<<dim3(BS / TILE_M, D / 128, 1), 256, 0, stream>>>(
        abf, D, 0, 0,  woT, D, 0, 0,  inter, D, 0, 0,
        bo, inputs, D, D, 1.0f, 1);

    // --- LN2 ---
    layernorm_bf16<<<BS, 256, 0, stream>>>(inter, ln2_g, ln2_b, x2bf, D);

    // --- MLP1: [4096,1024] x [1024,4096] + b1, exact GELU -> bf16 ---
    gemm_wmma_bf16<2, true, true, 128><<<dim3(BS / TILE_M, FF / 128, 1), 256, 0, stream>>>(
        x2bf, D, 0, 0,  w1T, D, 0, 0,  hbf, FF, 0, 0,
        b1, nullptr, 0, D, 1.0f, 1);

    // --- MLP2: [4096,4096] x [4096,1024] + b2 + residual(inter) -> out (f32) ---
    gemm_wmma_bf16<1, true, false, 128><<<dim3(BS / TILE_M, D / 128, 1), 256, 0, stream>>>(
        hbf, FF, 0, 0,  w2T, FF, 0, 0,  out, D, 0, 0,
        b2, inter, D, FF, 1.0f, 1);
}
